// ScaledDotProductAttention_2_86715389706696
// MI455X (gfx1250) — compile-verified, matrix operand order checked by reading, not run on hardware
//
#include <hip/hip_runtime.h>
#include <cstddef>

// Problem constants (from reference): B=2, H=4 -> BHN=8, L=2048, D=766
#define DK    766
#define LSEQ  2048
#define BHN   8
#define NEGV  -1000000000.0f

// GEMM tiling: workgroup = 256 threads = 8 waves (wave32), macro tile 64(M) x 128(N),
// K-step 32 (one bf16 WMMA depth). Each wave owns a 32x32 region = 4 16x16 accumulators.
#define TM   64
#define TN   128
#define TK   32
#define LDSS 40   // LDS row stride (ushorts): 16B-aligned chunks, staggered banks

typedef __attribute__((ext_vector_type(16))) __bf16          v16bf;
typedef __attribute__((ext_vector_type(8)))  float           v8f;
typedef __attribute__((ext_vector_type(8)))  unsigned short  ushort8;

union Frag16 { v16bf bf; ushort8 u[2]; };

__device__ __forceinline__ unsigned short f32_to_bf16_rne(float f) {
  unsigned int u = __float_as_uint(f);
  u += 0x7FFFu + ((u >> 16) & 1u);       // round-to-nearest-even
  return (unsigned short)(u >> 16);
}
__device__ __forceinline__ float bf16_hi_f32(unsigned short h) {
  return __uint_as_float(((unsigned int)h) << 16);
}
__device__ __forceinline__ void split_bf16(float f, unsigned short &hi, unsigned short &lo) {
  hi = f32_to_bf16_rne(f);
  lo = f32_to_bf16_rne(f - bf16_hi_f32(hi));   // residual -> near-fp32 via 3 products
}

// Load a 16x32 bf16 WMMA operand fragment from LDS tile `sub` (row-major, stride LDSS).
// CDNA5 16-bit A/B layout: lane<16 -> row=lane,   K = {0..7, 16..23}
//                          lane>=16 -> row=lane-16, K = {8..15, 24..31}
// i.e. two contiguous 8-element (16B) chunks per lane.
__device__ __forceinline__ v16bf load_frag(const unsigned short* sub, int lane) {
  Frag16 f;
  const unsigned short* p = sub + (lane & 15) * LDSS + ((lane >> 4) << 3);
  f.u[0] = *(const ushort8*)(p);
  f.u[1] = *(const ushort8*)(p + 16);
  return f.bf;
}

__device__ __forceinline__ v8f wmma_bf16(v16bf a, v16bf b, v8f c) {
  // (neg_a, A, neg_b, B, c_mod, C, reuse_a, reuse_b)
  return __builtin_amdgcn_wmma_f32_16x16x32_bf16(false, a, false, b, (short)0, c,
                                                 false, false);
}

// ---------------------------------------------------------------------------
// Stage 1: S[q,k] = (Q[q,:] . K[k,:]) / sqrt(DK), masked by attn_mask[b,h,k,q].
// Writes S into attn1 (row-major [q,k]) and S^T into attn2 ([k,q]).
// ---------------------------------------------------------------------------
__global__ __launch_bounds__(256) void qk_scores_kernel(
    const float* __restrict__ Q, const float* __restrict__ Kmat,
    const unsigned char* __restrict__ mask,
    float* __restrict__ attn1, float* __restrict__ attn2)
{
  __shared__ __align__(16) unsigned short a_hi[TM * LDSS], a_lo[TM * LDSS];
  __shared__ __align__(16) unsigned short b_hi[TN * LDSS], b_lo[TN * LDSS];

  const int bh     = blockIdx.z;
  const int m_base = blockIdx.y * TM;   // q
  const int n_base = blockIdx.x * TN;   // k
  const int tid    = threadIdx.x;
  const int lane   = tid & 31;
  const int wave   = tid >> 5;
  const int wm     = (wave & 1) * 32;   // 2 wave-rows
  const int wn     = (wave >> 1) * 32;  // 4 wave-cols

  const float* Qp = Q    + (size_t)bh * LSEQ * DK;
  const float* Kp = Kmat + (size_t)bh * LSEQ * DK;

  v8f acc[2][2];
  #pragma unroll
  for (int i = 0; i < 2; ++i)
    #pragma unroll
    for (int j = 0; j < 2; ++j)
      #pragma unroll
      for (int e = 0; e < 8; ++e) acc[i][j][e] = 0.0f;

  for (int k0 = 0; k0 < DK; k0 += TK) {
    // Stage A tile (Q): TM x TK, fp32 -> bf16 hi/lo, zero-pad K tail
    for (int idx = tid; idx < TM * TK; idx += 256) {
      const int r = idx >> 5, c = idx & 31;
      const int gk = k0 + c;
      const float f = (gk < DK) ? Qp[(size_t)(m_base + r) * DK + gk] : 0.0f;
      unsigned short hi, lo; split_bf16(f, hi, lo);
      a_hi[r * LDSS + c] = hi;  a_lo[r * LDSS + c] = lo;
    }
    // Stage B tile (K rows along d; WMMA-B column n == K row n): TN x TK
    for (int idx = tid; idx < TN * TK; idx += 256) {
      const int r = idx >> 5, c = idx & 31;
      const int gk = k0 + c;
      const float f = (gk < DK) ? Kp[(size_t)(n_base + r) * DK + gk] : 0.0f;
      unsigned short hi, lo; split_bf16(f, hi, lo);
      b_hi[r * LDSS + c] = hi;  b_lo[r * LDSS + c] = lo;
    }
    __syncthreads();

    v16bf ah[2], al[2], bhv[2], blv[2];
    #pragma unroll
    for (int s = 0; s < 2; ++s) {
      ah[s]  = load_frag(a_hi + (wm + s * 16) * LDSS, lane);
      al[s]  = load_frag(a_lo + (wm + s * 16) * LDSS, lane);
      bhv[s] = load_frag(b_hi + (wn + s * 16) * LDSS, lane);
      blv[s] = load_frag(b_lo + (wn + s * 16) * LDSS, lane);
    }
    #pragma unroll
    for (int i = 0; i < 2; ++i)
      #pragma unroll
      for (int j = 0; j < 2; ++j) {
        acc[i][j] = wmma_bf16(ah[i], bhv[j], acc[i][j]);  // hi*hi
        acc[i][j] = wmma_bf16(ah[i], blv[j], acc[i][j]);  // hi*lo
        acc[i][j] = wmma_bf16(al[i], bhv[j], acc[i][j]);  // lo*hi
      }
    __syncthreads();
  }

  // Epilogue: scale, mask (mask is [b,h,k,q]), dual store S and S^T
  const float scale = 0.03613130432f;  // 1/sqrt(766)
  float* A1 = attn1 + (size_t)bh * LSEQ * LSEQ;
  float* A2 = attn2 + (size_t)bh * LSEQ * LSEQ;
  const unsigned char* Mk = mask + (size_t)bh * LSEQ * LSEQ;
  #pragma unroll
  for (int i = 0; i < 2; ++i)
    #pragma unroll
    for (int j = 0; j < 2; ++j) {
      const int q0 = m_base + wm + i * 16 + ((lane >> 4) << 3);
      const int k  = n_base + wn + j * 16 + (lane & 15);
      #pragma unroll
      for (int r = 0; r < 8; ++r) {
        const int q = q0 + r;
        float s = acc[i][j][r] * scale;
        if (Mk[(size_t)k * LSEQ + q]) s = NEGV;
        A1[(size_t)q * LSEQ + k] = s;
        A2[(size_t)k * LSEQ + q] = s;
      }
    }
}

// ---------------------------------------------------------------------------
// Stage 2: in-place row softmax over the last axis (row length LSEQ=2048).
// One 256-thread block per row, 8 elements per thread.
// ---------------------------------------------------------------------------
__global__ __launch_bounds__(256) void softmax_rows_kernel(float* __restrict__ buf)
{
  float* p = buf + (size_t)blockIdx.x * LSEQ;
  const int tid = threadIdx.x;
  __shared__ float red[256];

  float x[8];
  float m = -3.4e38f;
  #pragma unroll
  for (int i = 0; i < 8; ++i) { x[i] = p[tid + i * 256]; m = fmaxf(m, x[i]); }
  red[tid] = m; __syncthreads();
  for (int s = 128; s > 0; s >>= 1) {
    if (tid < s) red[tid] = fmaxf(red[tid], red[tid + s]);
    __syncthreads();
  }
  m = red[0]; __syncthreads();

  float sum = 0.0f;
  #pragma unroll
  for (int i = 0; i < 8; ++i) { x[i] = __expf(x[i] - m); sum += x[i]; }
  red[tid] = sum; __syncthreads();
  for (int s = 128; s > 0; s >>= 1) {
    if (tid < s) red[tid] += red[tid + s];
    __syncthreads();
  }
  const float inv = 1.0f / red[0];
  #pragma unroll
  for (int i = 0; i < 8; ++i) p[tid + i * 256] = x[i] * inv;
}

// ---------------------------------------------------------------------------
// Stage 3: Out[m,d] = sum_k P[m,k] * V[k,d]  (P: [L,L] probs, V: [L,DK]).
// B (V) is d-major in memory; transpose into k-major LDS during staging.
// ---------------------------------------------------------------------------
__global__ __launch_bounds__(256) void pv_context_kernel(
    const float* __restrict__ P, const float* __restrict__ V,
    float* __restrict__ Out)
{
  __shared__ __align__(16) unsigned short a_hi[TM * LDSS], a_lo[TM * LDSS];
  __shared__ __align__(16) unsigned short b_hi[TN * LDSS], b_lo[TN * LDSS];

  const int bh     = blockIdx.z;
  const int m_base = blockIdx.y * TM;   // row (q or k)
  const int n_base = blockIdx.x * TN;   // d
  const int tid    = threadIdx.x;
  const int lane   = tid & 31;
  const int wave   = tid >> 5;
  const int wm     = (wave & 1) * 32;
  const int wn     = (wave >> 1) * 32;

  const float* Pp = P + (size_t)bh * LSEQ * LSEQ;
  const float* Vp = V + (size_t)bh * LSEQ * DK;

  v8f acc[2][2];
  #pragma unroll
  for (int i = 0; i < 2; ++i)
    #pragma unroll
    for (int j = 0; j < 2; ++j)
      #pragma unroll
      for (int e = 0; e < 8; ++e) acc[i][j][e] = 0.0f;

  for (int k0 = 0; k0 < LSEQ; k0 += TK) {
    // Stage A tile (probabilities, k-contiguous): TM x TK, no K guard (2048%32==0)
    for (int idx = tid; idx < TM * TK; idx += 256) {
      const int r = idx >> 5, c = idx & 31;
      const float f = Pp[(size_t)(m_base + r) * LSEQ + (k0 + c)];
      unsigned short hi, lo; split_bf16(f, hi, lo);
      a_hi[r * LDSS + c] = hi;  a_lo[r * LDSS + c] = lo;
    }
    // Stage B tile transposed: b[n][c] = V[k0+c][n_base+n]; consecutive tid -> consecutive d
    for (int idx = tid; idx < TN * TK; idx += 256) {
      const int r = idx & 127;      // n (d) local
      const int c = idx >> 7;       // k local
      const int gd = n_base + r;
      const float f = (gd < DK) ? Vp[(size_t)(k0 + c) * DK + gd] : 0.0f;
      unsigned short hi, lo; split_bf16(f, hi, lo);
      b_hi[r * LDSS + c] = hi;  b_lo[r * LDSS + c] = lo;
    }
    __syncthreads();

    v16bf ah[2], al[2], bhv[2], blv[2];
    #pragma unroll
    for (int s = 0; s < 2; ++s) {
      ah[s]  = load_frag(a_hi + (wm + s * 16) * LDSS, lane);
      al[s]  = load_frag(a_lo + (wm + s * 16) * LDSS, lane);
      bhv[s] = load_frag(b_hi + (wn + s * 16) * LDSS, lane);
      blv[s] = load_frag(b_lo + (wn + s * 16) * LDSS, lane);
    }
    #pragma unroll
    for (int i = 0; i < 2; ++i)
      #pragma unroll
      for (int j = 0; j < 2; ++j) {
        acc[i][j] = wmma_bf16(ah[i], bhv[j], acc[i][j]);
        acc[i][j] = wmma_bf16(ah[i], blv[j], acc[i][j]);
        acc[i][j] = wmma_bf16(al[i], bhv[j], acc[i][j]);
      }
    __syncthreads();
  }

  float* Op = Out + (size_t)bh * LSEQ * DK;
  #pragma unroll
  for (int i = 0; i < 2; ++i)
    #pragma unroll
    for (int j = 0; j < 2; ++j) {
      const int m0 = m_base + wm + i * 16 + ((lane >> 4) << 3);
      const int d  = n_base + wn + j * 16 + (lane & 15);
      if (d < DK) {
        #pragma unroll
        for (int r = 0; r < 8; ++r)
          Op[(size_t)(m0 + r) * DK + d] = acc[i][j][r];
      }
    }
}

// ---------------------------------------------------------------------------
extern "C" void kernel_launch(void* const* d_in, const int* in_sizes, int n_in,
                              void* d_out, int out_size, void* d_ws, size_t ws_size,
                              hipStream_t stream) {
  (void)in_sizes; (void)n_in; (void)out_size; (void)d_ws; (void)ws_size;

  const float*         Q    = (const float*)d_in[0];
  const float*         Kmat = (const float*)d_in[1];
  const unsigned char* mask = (const unsigned char*)d_in[2];  // numpy bool, 1 byte

  // Output layout (flat, in tuple order): context1, context2, attn1, attn2
  const size_t ctx_elems  = (size_t)BHN * LSEQ * DK;     // 12,550,144
  const size_t attn_elems = (size_t)BHN * LSEQ * LSEQ;   // 33,554,432
  float* out   = (float*)d_out;
  float* ctx1  = out;
  float* ctx2  = out + ctx_elems;
  float* attn1 = out + 2 * ctx_elems;
  float* attn2 = out + 2 * ctx_elems + attn_elems;

  const dim3 blk(256);

  // 1) scores + scale + mask -> attn1 (S) and attn2 (S^T)
  qk_scores_kernel<<<dim3(LSEQ / TN, LSEQ / TM, BHN), blk, 0, stream>>>(
      Q, Kmat, mask, attn1, attn2);

  // 2) in-place row softmax on both score buffers
  softmax_rows_kernel<<<dim3(BHN * LSEQ), blk, 0, stream>>>(attn1);
  softmax_rows_kernel<<<dim3(BHN * LSEQ), blk, 0, stream>>>(attn2);

  // 3) contexts: ctx1 = attn1 @ K, ctx2 = attn2 @ Q
  const int ntiles = (DK + TN - 1) / TN;  // 6
  pv_context_kernel<<<dim3(ntiles, LSEQ / TM, BHN), blk, 0, stream>>>(attn1, Kmat, ctx1);
  pv_context_kernel<<<dim3(ntiles, LSEQ / TM, BHN), blk, 0, stream>>>(attn2, Q, ctx2);
}
